// SwitchedConvHardRouting_4423816315014
// MI455X (gfx1250) — matrix operand order repaired and checked by reading, not compiled, not run on hardware
//
#include <hip/hip_runtime.h>
#include <hip/hip_bf16.h>

typedef __bf16 bf16_t;
typedef __attribute__((ext_vector_type(8)))  __bf16 v8bf;
typedef __attribute__((ext_vector_type(16))) __bf16 v16bf;
typedef __attribute__((ext_vector_type(8)))  float  v8f;

#define B_    4
#define CIN_  128
#define COUT_ 128
#define S_    8
#define H_    128
#define W_    128
#define KK    (CIN_ * 9)          // 1152 = GEMM K dimension
#define LDS_STRIDE 1168           // padded K stride; 2336 B = 73*32 (keeps 32B align)

#define W_ELEMS  (S_ * COUT_ * KK)   // 1,179,648 bf16 weights in ws
#define MASK_OFF ((size_t)W_ELEMS * 2)

// ---- prep 1: weight [Cout,Cin,S,3,3] f32 -> [S][Cout][Cin*9] bf16 ----------
__global__ void convert_weights(const float* __restrict__ w, bf16_t* __restrict__ wb) {
  int i = blockIdx.x * 256 + threadIdx.x;
  if (i >= W_ELEMS) return;
  int k   = i % KK;
  int row = i / KK;            // s*COUT + co
  int co  = row % COUT_;
  int s   = row / COUT_;
  int cin = k / 9;
  int r   = k - cin * 9;       // ky*3+kx
  int src = (((co * CIN_ + cin) * S_ + s) * 9) + r;
  wb[i] = (bf16_t)w[src];
}

// ---- prep 2: per-pixel argmax over S experts -------------------------------
__global__ void compute_mask(const float* __restrict__ sel, unsigned char* __restrict__ mask) {
  int i = blockIdx.x * 256 + threadIdx.x;   // b*H*W + h*W + w
  if (i >= B_ * H_ * W_) return;
  int b  = i / (H_ * W_);
  int hw = i - b * (H_ * W_);
  float best = sel[(b * S_) * H_ * W_ + hw];
  int bi = 0;
  for (int s = 1; s < S_; ++s) {
    float v = sel[(b * S_ + s) * H_ * W_ + hw];
    if (v > best) { best = v; bi = s; }
  }
  mask[i] = (unsigned char)bi;
}

// ---- main: implicit-GEMM switched conv, bf16 WMMA, f32 accumulate ----------
// 128 threads = 4 waves; each wave owns TWO cout tiles (co, co+64) that share
// one LDS B-fragment per K-chunk -> 2 independent WMMA chains per wave.
__global__ __launch_bounds__(128)
void switched_conv_wmma(const float* __restrict__ x, const bf16_t* __restrict__ wb,
                        const unsigned char* __restrict__ mask,
                        const float* __restrict__ bias, float* __restrict__ out) {
  __shared__ __align__(32) bf16_t xp[16 * LDS_STRIDE];  // B matrix: 16 pixels x K
  __shared__ unsigned presentMask;
  __shared__ unsigned char smask[16];

  const int w0  = blockIdx.x * 16;
  const int h   = blockIdx.y;
  const int b   = blockIdx.z;
  const int tid = threadIdx.x;

  if (tid == 0) presentMask = 0;
  __syncthreads();
  if (tid < 16) {
    unsigned char m = mask[(b * H_ + h) * W_ + w0 + tid];
    smask[tid] = m;
    atomicOr(&presentMask, 1u << m);
  }

  // im2col patch into LDS: xp[n][k], k = cin*9 + ky*3 + kx, 'same' padding
  for (int idx = tid; idx < 16 * KK; idx += 128) {
    int n   = idx / KK;
    int k   = idx - n * KK;
    int cin = k / 9;
    int r   = k - cin * 9;
    int ky  = r / 3, kx = r - ky * 3;
    int hh  = h + ky - 1;
    int ww  = w0 + n + kx - 1;
    float v = 0.f;
    if (hh >= 0 && hh < H_ && ww >= 0 && ww < W_)
      v = x[((b * CIN_ + cin) * H_ + hh) * W_ + ww];
    xp[n * LDS_STRIDE + k] = (bf16_t)v;
  }
  __syncthreads();

  const int lane = tid & 31;
  const int wave = tid >> 5;          // 4 waves
  const int coA  = wave * 16;         // first cout tile
  const int coB  = coA + 64;          // second cout tile (shares B fragment)
  const int m    = lane & 15;         // A: row M / B,C,D: column N
  const int hi   = lane >> 4;
  const unsigned pm = (unsigned)__builtin_amdgcn_readfirstlane((int)presentMask);
  const int myExpert = smask[m];      // expert of this lane's output column

  const bf16_t* ldsB   = &xp[m * LDS_STRIDE + 8 * hi]; // K = {0..7|8..15} and {16..23|24..31} per lane
  const bf16_t* aBaseA = wb + (size_t)(coA + m) * KK;
  const bf16_t* aBaseB = wb + (size_t)(coB + m) * KK;
  const int koffA = hi ? 8 : 0;       // A K-group layout per ISA table

  v8f dselA, dselB;
#pragma unroll
  for (int r = 0; r < 8; ++r) { dselA[r] = 0.f; dselB[r] = 0.f; }

  for (int s = 0; s < S_; ++s) {
    if (!(pm & (1u << s))) continue;                 // wave-uniform skip
    const bf16_t* arowA = aBaseA + (size_t)s * (COUT_ * KK);
    const bf16_t* arowB = aBaseB + (size_t)s * (COUT_ * KK);
    v8f cA, cB;
#pragma unroll
    for (int r = 0; r < 8; ++r) { cA[r] = 0.f; cB[r] = 0.f; }
    for (int kc = 0; kc < KK; kc += 32) {
      __builtin_prefetch((const void*)(arowA + kc + 64), 0, 0);
      __builtin_prefetch((const void*)(arowB + kc + 64), 0, 0);
      v8bf aA_lo = *(const v8bf*)(arowA + kc + koffA);      // K = kc+{0..7 | 8..15}
      v8bf aA_hi = *(const v8bf*)(arowA + kc + 16 + koffA); // K = kc+{16..23 | 24..31}
      v8bf aB_lo = *(const v8bf*)(arowB + kc + koffA);
      v8bf aB_hi = *(const v8bf*)(arowB + kc + 16 + koffA);
      v16bf aA = __builtin_shufflevector(aA_lo, aA_hi,
                                         0,1,2,3,4,5,6,7,8,9,10,11,12,13,14,15);
      v16bf aB = __builtin_shufflevector(aB_lo, aB_hi,
                                         0,1,2,3,4,5,6,7,8,9,10,11,12,13,14,15);
      v16bf bfrag = __builtin_shufflevector(*(const v8bf*)(ldsB + kc), *(const v8bf*)(ldsB + kc + 16),
                                            0,1,2,3,4,5,6,7,8,9,10,11,12,13,14,15); // shared by both tiles
      cA = __builtin_amdgcn_wmma_f32_16x16x32_bf16(false, aA, false, bfrag,
                                                   (short)0, cA, false, false);
      cB = __builtin_amdgcn_wmma_f32_16x16x32_bf16(false, aB, false, bfrag,
                                                   (short)0, cB, false, false);
    }
    if (myExpert == s) { dselA = cA; dselB = cB; }  // per-pixel hard routing
  }

#pragma unroll
  for (int r = 0; r < 8; ++r) {
    int cA_ = coA + r + 8 * hi;                     // C/D VGPR layout: M = r + 8*hi
    int cB_ = coB + r + 8 * hi;
    out[((b * COUT_ + cA_) * H_ + h) * W_ + w0 + m] = dselA[r] + bias[cA_];
    out[((b * COUT_ + cB_) * H_ + h) * W_ + w0 + m] = dselB[r] + bias[cB_];
  }
}

extern "C" void kernel_launch(void* const* d_in, const int* in_sizes, int n_in,
                              void* d_out, int out_size, void* d_ws, size_t ws_size,
                              hipStream_t stream) {
  (void)in_sizes; (void)n_in; (void)out_size; (void)ws_size;
  const float* x    = (const float*)d_in[0];
  const float* sel  = (const float*)d_in[1];
  const float* w    = (const float*)d_in[2];
  const float* bias = (const float*)d_in[3];
  float* out = (float*)d_out;

  bf16_t* wb = (bf16_t*)d_ws;
  unsigned char* mask = (unsigned char*)((char*)d_ws + MASK_OFF);

  convert_weights<<<(W_ELEMS + 255) / 256, 256, 0, stream>>>(w, wb);
  compute_mask<<<(B_ * H_ * W_ + 255) / 256, 256, 0, stream>>>(sel, mask);

  dim3 grid(W_ / 16, H_, B_);
  switched_conv_wmma<<<grid, 128, 0, stream>>>(x, wb, mask, bias, out);
}
